// SimpleTGN_37460704755813
// MI455X (gfx1250) — matrix operand order repaired and checked by reading, not compiled, not run on hardware
//
#include <hip/hip_runtime.h>
#include <hip/hip_bf16.h>

// Problem constants from the reference
constexpr int kN  = 100000;
constexpr int kE  = 1000000;
constexpr int kND = 128;
constexpr int kED = 64;
constexpr int kH  = 64;
constexpr int kP  = 10000;

constexpr int kEdgeTiles  = kE / 16;   // 62500
constexpr int kEdgeBlocks = 2048;      // grid-stride: ~3.8 tiles per wave

typedef __bf16 bf16_t;
typedef __attribute__((ext_vector_type(16))) __bf16 v16bf;
typedef __attribute__((ext_vector_type(8)))  __bf16 v8bf;
typedef __attribute__((ext_vector_type(4)))  __bf16 v4bf;
typedef __attribute__((ext_vector_type(8)))  float  v8f;
typedef __attribute__((ext_vector_type(4)))  float  v4f;

static __device__ __forceinline__ v8f wmma_bf16(v16bf a, v16bf b, v8f c) {
  // (neg_a, A, neg_b, B, c_mod, C, reuse_a, reuse_b)
  return __builtin_amdgcn_wmma_f32_16x16x32_bf16(false, a, false, b, (short)0, c,
                                                 false, false);
}

// Wave-local LDS producer->consumer ordering (same wave, cross-lane):
// DS ops are in-order per wave; the asm keeps the compiler from reordering.
static __device__ __forceinline__ void lds_sync_wave() {
  asm volatile("s_wait_dscnt 0" ::: "memory");
}

// A fragment (16x32, bf16) from a row-major f32 matrix. `row` points at this
// lane's matrix row (m = lane&15). Lanes 0-15 hold K = kbase+[0..7]
// (elems 0..7) and kbase+16+[0..7] (elems 8..15); lanes 16-31 hold +8.
static __device__ __forceinline__ v16bf load_a_from_f32(const float* row,
                                                        int kbase, int half) {
  const v4f* p = (const v4f*)(row + kbase + half * 8);
  v4f x0 = p[0], x1 = p[1], x2 = p[4], x3 = p[5];
  v16bf a;
#pragma unroll
  for (int i = 0; i < 4; ++i) {
    a[i]      = (bf16_t)x0[i];
    a[4 + i]  = (bf16_t)x1[i];
    a[8 + i]  = (bf16_t)x2[i];
    a[12 + i] = (bf16_t)x3[i];
  }
  return a;
}

// Same fragment, sourced from bf16 (global or LDS; 16B-aligned runs).
static __device__ __forceinline__ v16bf load_a_from_bf16(const bf16_t* row,
                                                         int kbase, int half) {
  const v8bf* p = (const v8bf*)(row + kbase + half * 8);
  v8bf x0 = p[0];
  v8bf x1 = p[2];  // +16 bf16 elements
  v16bf a;
#pragma unroll
  for (int i = 0; i < 8; ++i) { a[i] = x0[i]; a[8 + i] = x1[i]; }
  return a;
}

// B fragment (32x16, bf16) from a PREPACKED row-major bf16 weight Wb[out][in]:
// B[k][n] = Wb[ncol_base + lane&15][k]; lane reads K = kb+[0..7] and
// kb+16+[0..7], kb = kbase + (lane>>4)*8 -> two 16B loads, no conversion.
static __device__ __forceinline__ v16bf load_b_bf16(const bf16_t* Wb, int ldw,
                                                    int ncol_base, int kbase,
                                                    int lane) {
  int n  = ncol_base + (lane & 15);
  int kb = kbase + (lane >> 4) * 8;
  const v8bf* p = (const v8bf*)(Wb + (size_t)n * ldw + kb);
  v8bf x0 = p[0], x1 = p[2];
  v16bf b;
#pragma unroll
  for (int i = 0; i < 8; ++i) { b[i] = x0[i]; b[8 + i] = x1[i]; }
  return b;
}

static __device__ __forceinline__ v8f splat8(float x) {
  v8f r;
#pragma unroll
  for (int i = 0; i < 8; ++i) r[i] = x;
  return r;
}

static __device__ __forceinline__ float fast_sigmoid(float x) {
  return 1.0f / (1.0f + __expf(-x));
}
static __device__ __forceinline__ float fast_tanh(float x) {
  float t = __expf(-2.0f * x);
  return (1.0f - t) / (1.0f + t);
}

// ---------------------------------------------------------------------------
__global__ __launch_bounds__(256) void k_zero(float* __restrict__ p, int n4) {
  int i = blockIdx.x * 256 + threadIdx.x;
  v4f z = {0.f, 0.f, 0.f, 0.f};
  for (; i < n4; i += gridDim.x * 256) ((v4f*)p)[i] = z;
}

// Pack an f32 array to bf16 (n divisible by 4).
__global__ __launch_bounds__(256) void k_pack(const float* __restrict__ src,
                                              bf16_t* __restrict__ dst, int n4) {
  int i = blockIdx.x * 256 + threadIdx.x;
  for (; i < n4; i += gridDim.x * 256) {
    v4f x = ((const v4f*)src)[i];
    v4bf y;
#pragma unroll
    for (int j = 0; j < 4; ++j) y[j] = (bf16_t)x[j];
    ((v4bf*)dst)[i] = y;
  }
}

// node_emb = relu(NF @ W_ne^T + b_ne); writes f32 + bf16 copies.
__global__ __launch_bounds__(256) void k_node_emb(
    const float* __restrict__ nf, const bf16_t* __restrict__ Wne,
    const float* __restrict__ bne, float* __restrict__ node_emb,
    bf16_t* __restrict__ node_emb_bf) {
  int wave = (blockIdx.x << 3) + (threadIdx.x >> 5);
  if (wave >= kN / 16) return;           // wave-uniform: EXEC all-ones below
  int lane = threadIdx.x & 31;
  int half = lane >> 4;
  int nl   = lane & 15;
  int r0   = wave * 16;

  const float* arow = nf + (size_t)(r0 + nl) * kND;
  v16bf af[4];
#pragma unroll
  for (int k = 0; k < 4; ++k) af[k] = load_a_from_f32(arow, k * 32, half);

#pragma unroll
  for (int nt = 0; nt < 4; ++nt) {
    int col = nt * 16 + nl;
    v8f acc = {};                        // inline-0 C on first WMMA
#pragma unroll
    for (int k = 0; k < 4; ++k) {
      v16bf b = load_b_bf16(Wne, kND, nt * 16, k * 32, lane);
      acc = wmma_bf16(af[k], b, acc);
    }
    float bias = bne[col];
#pragma unroll
    for (int v = 0; v < 8; ++v) {
      float x = fmaxf(acc[v] + bias, 0.0f);
      size_t idx = (size_t)(r0 + v + 8 * half) * kH + col;
      node_emb[idx]    = x;
      node_emb_bf[idx] = (bf16_t)x;
    }
  }
}

// Fused per-edge pipeline, grid-stride over 16-edge tiles with the 16 W_m
// B fragments held in registers across tiles (amortizes the dominant weight
// fetch). Per tile: edge_emb GEMM -> LDS bf16 tile; gather node_emb_bf[src];
// msg GEMM over concat; atomic scatter-add to messages[dst].
__global__ __launch_bounds__(256) void k_edges(
    const float* __restrict__ ea, const int* __restrict__ ei,
    const bf16_t* __restrict__ Wee, const float* __restrict__ bee,
    const bf16_t* __restrict__ Wm, const float* __restrict__ bm,
    const bf16_t* __restrict__ node_emb_bf, float* __restrict__ messages) {
  __shared__ bf16_t smem[8][16][kH];     // per-wave 16x64 bf16 edge_emb tile
  int w    = threadIdx.x >> 5;
  int lane = threadIdx.x & 31;
  int half = lane >> 4;
  int nl   = lane & 15;
  int wid    = (blockIdx.x << 3) + w;    // wave-uniform
  int stride = kEdgeBlocks << 3;

  // Persistent W_m fragments: [nt][k] with k 0..1 = node half (cols 0..63),
  // k 2..3 = edge half (cols 64..127). 16 x v16bf = 128 VGPRs.
  v16bf wm[4][4];
#pragma unroll
  for (int nt = 0; nt < 4; ++nt)
#pragma unroll
    for (int k = 0; k < 4; ++k)
      wm[nt][k] = load_b_bf16(Wm, 2 * kH, nt * 16,
                              (k < 2) ? k * 32 : kH + (k - 2) * 32, lane);

  for (int tile = wid; tile < kEdgeTiles; tile += stride) {  // wave-uniform
    int e0 = tile * 16;

    // ---- edge_emb = relu(edge_attr @ Wee^T + bee) -> LDS (bf16) ----
    const float* arow = ea + (size_t)(e0 + nl) * kED;
    __builtin_prefetch(arow + 16 * kED, 0, 1);   // next tile's stream
    v16bf aeatt[2];
#pragma unroll
    for (int k = 0; k < 2; ++k) aeatt[k] = load_a_from_f32(arow, k * 32, half);

#pragma unroll
    for (int nt = 0; nt < 4; ++nt) {
      v8f acc = {};
#pragma unroll
      for (int k = 0; k < 2; ++k) {
        v16bf b = load_b_bf16(Wee, kED, nt * 16, k * 32, lane);
        acc = wmma_bf16(aeatt[k], b, acc);
      }
      float bias = bee[nt * 16 + nl];
#pragma unroll
      for (int v = 0; v < 8; ++v) {
        float x = fmaxf(acc[v] + bias, 0.0f);
        smem[w][v + 8 * half][nt * 16 + nl] = (bf16_t)x;
      }
    }
    lds_sync_wave();

    // ---- gather indices ----
    int s = ei[e0 + nl];                 // src row for this lane's A row
    const bf16_t* srow = node_emb_bf + (size_t)s * kH;
    int d_own = ei[kE + e0 + nl];        // lane nl holds dst of edge row nl
    int d[8];
#pragma unroll
    for (int v = 0; v < 8; ++v)          // broadcast rows v+8*half via shuffle
      d[v] = __shfl(d_own, v + 8 * half, 32);

    v16bf an[2], aeg[2];
#pragma unroll
    for (int k = 0; k < 2; ++k) {
      an[k]  = load_a_from_bf16(srow, k * 32, half);             // gather
      aeg[k] = load_a_from_bf16(&smem[w][nl][0], k * 32, half);  // LDS tile
    }

    // ---- msg = [n_src | e_emb] @ Wm^T + bm ; scatter-add ----
#pragma unroll
    for (int nt = 0; nt < 4; ++nt) {
      int col = nt * 16 + nl;
      v8f acc = splat8(bm[col]);         // b_m added once per edge (reference)
      acc = wmma_bf16(an[0],  wm[nt][0], acc);
      acc = wmma_bf16(an[1],  wm[nt][1], acc);
      acc = wmma_bf16(aeg[0], wm[nt][2], acc);
      acc = wmma_bf16(aeg[1], wm[nt][3], acc);
#pragma unroll
      for (int v = 0; v < 8; ++v)
        atomicAdd(&messages[(size_t)d[v] * kH + col], acc[v]);
    }
  }
}

// GRUCell(messages, node_emb) -> new_h
__global__ __launch_bounds__(256) void k_gru(
    const float* __restrict__ messages, const float* __restrict__ node_emb,
    const bf16_t* __restrict__ node_emb_bf, const bf16_t* __restrict__ Wih,
    const float* __restrict__ bih, const bf16_t* __restrict__ Whh,
    const float* __restrict__ bhh, float* __restrict__ new_h) {
  int wave = (blockIdx.x << 3) + (threadIdx.x >> 5);
  if (wave >= kN / 16) return;           // wave-uniform
  int lane = threadIdx.x & 31;
  int half = lane >> 4;
  int nl   = lane & 15;
  int r0   = wave * 16;

  const float*  mrow = messages + (size_t)(r0 + nl) * kH;
  const bf16_t* hrow = node_emb_bf + (size_t)(r0 + nl) * kH;
  v16bf am[2], ah[2];
#pragma unroll
  for (int k = 0; k < 2; ++k) {
    am[k] = load_a_from_f32(mrow, k * 32, half);
    ah[k] = load_a_from_bf16(hrow, k * 32, half);
  }

#pragma unroll
  for (int j = 0; j < 4; ++j) {
    int col = j * 16 + nl;
    v8f ir = {}, iz = {}, in_ = {}, hr = {}, hz = {}, hn = {};  // inline-0 C
#pragma unroll
    for (int k = 0; k < 2; ++k) {
      v16bf br = load_b_bf16(Wih,            kH, j * 16, k * 32, lane);
      v16bf bz = load_b_bf16(Wih +  64 * kH, kH, j * 16, k * 32, lane);
      v16bf bn = load_b_bf16(Wih + 128 * kH, kH, j * 16, k * 32, lane);
      ir  = wmma_bf16(am[k], br, ir);
      iz  = wmma_bf16(am[k], bz, iz);
      in_ = wmma_bf16(am[k], bn, in_);
      br = load_b_bf16(Whh,            kH, j * 16, k * 32, lane);
      bz = load_b_bf16(Whh +  64 * kH, kH, j * 16, k * 32, lane);
      bn = load_b_bf16(Whh + 128 * kH, kH, j * 16, k * 32, lane);
      hr = wmma_bf16(ah[k], br, hr);
      hz = wmma_bf16(ah[k], bz, hz);
      hn = wmma_bf16(ah[k], bn, hn);
    }
    // r/z gates see gi+gh, so their two biases combine into one scalar.
    float br_c = bih[col] + bhh[col];
    float bz_c = bih[64 + col] + bhh[64 + col];
    float bin  = bih[128 + col];
    float bhn  = bhh[128 + col];
#pragma unroll
    for (int v = 0; v < 8; ++v) {
      size_t idx = (size_t)(r0 + v + 8 * half) * kH + col;
      float hprev = node_emb[idx];
      float r  = fast_sigmoid(ir[v] + hr[v] + br_c);
      float z  = fast_sigmoid(iz[v] + hz[v] + bz_c);
      float nn = fast_tanh(in_[v] + bin + r * (hn[v] + bhn));
      new_h[idx] = (1.0f - z) * nn + z * hprev;
    }
  }
}

// out = sigmoid(new_h[post_mask] @ W_o^T + b_o)
__global__ __launch_bounds__(256) void k_out(const float* __restrict__ new_h,
                                             const int* __restrict__ post_mask,
                                             const float* __restrict__ Wo,
                                             const float* __restrict__ bo,
                                             float* __restrict__ out) {
  int i = blockIdx.x * 256 + threadIdx.x;
  if (i >= kP) return;
  int p = post_mask[i];
  const v4f* h = (const v4f*)(new_h + (size_t)p * kH);
  const v4f* w = (const v4f*)Wo;
  float s = bo[0];
#pragma unroll
  for (int g = 0; g < 16; ++g) {
    v4f a = h[g], b = w[g];
    s += a[0] * b[0] + a[1] * b[1] + a[2] * b[2] + a[3] * b[3];
  }
  out[i] = fast_sigmoid(s);
}

// ---------------------------------------------------------------------------
extern "C" void kernel_launch(void* const* d_in, const int* in_sizes, int n_in,
                              void* d_out, int out_size, void* d_ws,
                              size_t ws_size, hipStream_t stream) {
  (void)in_sizes; (void)n_in; (void)out_size; (void)ws_size;
  const float* nf  = (const float*)d_in[0];
  const int*   ei  = (const int*)d_in[1];
  const float* ea  = (const float*)d_in[2];
  const int*   pm  = (const int*)d_in[3];
  const float* Wne = (const float*)d_in[4];
  const float* bne = (const float*)d_in[5];
  const float* Wee = (const float*)d_in[6];
  const float* bee = (const float*)d_in[7];
  const float* Wm  = (const float*)d_in[8];
  const float* bm  = (const float*)d_in[9];
  const float* Wih = (const float*)d_in[10];
  const float* bih = (const float*)d_in[11];
  const float* Whh = (const float*)d_in[12];
  const float* bhh = (const float*)d_in[13];
  const float* Wo  = (const float*)d_in[14];
  const float* bo  = (const float*)d_in[15];
  float* out = (float*)d_out;

  // Workspace layout (ws poisoned -> zero what we accumulate into):
  float*  node_emb    = (float*)d_ws;
  float*  messages    = node_emb + (size_t)kN * kH;
  float*  new_h       = messages + (size_t)kN * kH;
  bf16_t* node_emb_bf = (bf16_t*)(new_h + (size_t)kN * kH);
  bf16_t* wbuf        = node_emb_bf + (size_t)kN * kH;
  bf16_t* Wne_b = wbuf;                  // 64 x 128
  bf16_t* Wee_b = Wne_b + kH * kND;      // 64 x 64
  bf16_t* Wm_b  = Wee_b + kH * kED;      // 64 x 128
  bf16_t* Wih_b = Wm_b + kH * 2 * kH;    // 192 x 64
  bf16_t* Whh_b = Wih_b + 3 * kH * kH;   // 192 x 64

  // Prepack weights to bf16 (tiny).
  k_pack<<<(kH * kND / 4 + 255) / 256, 256, 0, stream>>>(Wne, Wne_b, kH * kND / 4);
  k_pack<<<(kH * kED / 4 + 255) / 256, 256, 0, stream>>>(Wee, Wee_b, kH * kED / 4);
  k_pack<<<(kH * 2 * kH / 4 + 255) / 256, 256, 0, stream>>>(Wm, Wm_b, kH * 2 * kH / 4);
  k_pack<<<(3 * kH * kH / 4 + 255) / 256, 256, 0, stream>>>(Wih, Wih_b, 3 * kH * kH / 4);
  k_pack<<<(3 * kH * kH / 4 + 255) / 256, 256, 0, stream>>>(Whh, Whh_b, 3 * kH * kH / 4);

  k_zero<<<(kN * kH / 4 + 255) / 256, 256, 0, stream>>>(messages, kN * kH / 4);
  k_node_emb<<<(kN / 16 + 7) / 8, 256, 0, stream>>>(nf, Wne_b, bne, node_emb,
                                                    node_emb_bf);
  k_edges<<<kEdgeBlocks, 256, 0, stream>>>(ea, ei, Wee_b, bee, Wm_b, bm,
                                           node_emb_bf, messages);
  k_gru<<<(kN / 16 + 7) / 8, 256, 0, stream>>>(messages, node_emb, node_emb_bf,
                                               Wih_b, bih, Whh_b, bhh, new_h);
  k_out<<<(kP + 255) / 256, 256, 0, stream>>>(new_h, pm, Wo, bo, out);
}
